// StratifiedTransformerBackbone_1434519077263
// MI455X (gfx1250) — matrix operand order repaired and checked by reading, not compiled, not run on hardware
//
#include <hip/hip_runtime.h>
#include <hip/hip_bf16.h>

// ---------------------------------------------------------------------------
// Problem constants (from reference)
// ---------------------------------------------------------------------------
#define BB      8
#define N_IN    20000
#define N_ALL   1200
#define NPT     1024
#define KNB     34
#define CC      96
#define HH      6
#define DHD     16
#define ROWS    (BB * NPT)          // 8192
#define RADIUS2 0.01f               // (2.5*0.04)^2

typedef __attribute__((ext_vector_type(16))) _Float16 v16h;
typedef __attribute__((ext_vector_type(8)))  _Float16 v8h;
typedef __attribute__((ext_vector_type(8)))  float    v8f;

// ---------------------------------------------------------------------------
// 1) Farthest point sampling: one block per batch, 1024 threads.
//    Matches jax scan: emit current `far`, then update minDist, then argmax.
// ---------------------------------------------------------------------------
__global__ void fps_kernel(const float* __restrict__ xyz,
                           float* __restrict__ minDist,
                           int* __restrict__ fps_idx) {
    int b = blockIdx.x, tid = threadIdx.x;
    const float* pc = xyz + (size_t)b * N_IN * 3;
    float* md = minDist + (size_t)b * N_IN;
    int* fout = fps_idx + (size_t)b * N_ALL;

    __shared__ float sval[1024];
    __shared__ int   sidx[1024];
    __shared__ int   s_far;

    for (int i = tid; i < N_IN; i += 1024) md[i] = 1e10f;
    if (tid == 0) s_far = 0;
    __syncthreads();

    for (int n = 0; n < N_ALL; ++n) {
        int far = s_far;
        if (tid == 0) fout[n] = far;
        float cx = pc[far * 3 + 0], cy = pc[far * 3 + 1], cz = pc[far * 3 + 2];
        float best = -1.0f; int bidx = 0;
        for (int i = tid; i < N_IN; i += 1024) {
            float dx = pc[i * 3 + 0] - cx;
            float dy = pc[i * 3 + 1] - cy;
            float dz = pc[i * 3 + 2] - cz;
            float d = dx * dx + dy * dy + dz * dz;
            float m = fminf(md[i], d);
            md[i] = m;
            if (m > best) { best = m; bidx = i; }   // ascending i -> first max wins
        }
        sval[tid] = best; sidx[tid] = bidx;
        __syncthreads();
        for (int s = 512; s > 0; s >>= 1) {
            if (tid < s) {
                if (sval[tid + s] > sval[tid] ||
                    (sval[tid + s] == sval[tid] && sidx[tid + s] < sidx[tid])) {
                    sval[tid] = sval[tid + s]; sidx[tid] = sidx[tid + s];
                }
            }
            __syncthreads();
        }
        if (tid == 0) s_far = sidx[0];
        __syncthreads();
    }
}

// ---------------------------------------------------------------------------
// 2) Apply `sel`, gather coord/color; coord also goes to output tail.
// ---------------------------------------------------------------------------
__global__ void gather_kernel(const float* __restrict__ xyz,
                              const float* __restrict__ colors,
                              const int* __restrict__ fps_idx,
                              const int* __restrict__ sel,
                              float* __restrict__ coord,
                              float* __restrict__ color,
                              float* __restrict__ outCoord) {
    int t = blockIdx.x * blockDim.x + threadIdx.x;
    if (t >= ROWS) return;
    int b = t / NPT, p = t % NPT;
    int idx = fps_idx[b * N_ALL + sel[p]];
    #pragma unroll
    for (int j = 0; j < 3; ++j) {
        float cv = xyz[((size_t)b * N_IN + idx) * 3 + j];
        coord[(size_t)t * 3 + j] = cv;
        outCoord[(size_t)t * 3 + j] = cv;
        color[(size_t)t * 3 + j] = colors[((size_t)b * N_IN + idx) * 3 + j];
    }
}

// ---------------------------------------------------------------------------
// 3) kNN: one 256-thread block per (b,p). Coords staged in LDS.
//    34 stable min-selection rounds on (key, index), key = valid ? d2 : 1e10.
// ---------------------------------------------------------------------------
__global__ void knn_kernel(const float* __restrict__ coord,
                           int* __restrict__ nidx,
                           int* __restrict__ nvalid) {
    int blk = blockIdx.x;
    int b = blk / NPT, p = blk % NPT;
    int tid = threadIdx.x;

    __shared__ float sc[NPT * 3];
    __shared__ float key[NPT];
    __shared__ float rv[256];
    __shared__ int   ri[256];

    const float* cb = coord + (size_t)b * NPT * 3;
    for (int i = tid; i < NPT * 3; i += 256) sc[i] = cb[i];
    __syncthreads();

    float cx = sc[p * 3 + 0], cy = sc[p * 3 + 1], cz = sc[p * 3 + 2];
    for (int i = tid; i < NPT; i += 256) {
        float dx = sc[i * 3 + 0] - cx;
        float dy = sc[i * 3 + 1] - cy;
        float dz = sc[i * 3 + 2] - cz;
        float d = dx * dx + dy * dy + dz * dz;
        key[i] = (d <= RADIUS2) ? d : 1e10f;
    }
    __syncthreads();

    int* no = nidx + ((size_t)b * NPT + p) * KNB;
    int* vo = nvalid + ((size_t)b * NPT + p) * KNB;

    for (int k = 0; k < KNB; ++k) {
        float best = 3e38f; int bi = NPT;
        for (int i = tid; i < NPT; i += 256) {
            float kv = key[i];
            if (kv < best || (kv == best && i < bi)) { best = kv; bi = i; }
        }
        rv[tid] = best; ri[tid] = bi;
        __syncthreads();
        for (int s = 128; s > 0; s >>= 1) {
            if (tid < s) {
                if (rv[tid + s] < rv[tid] ||
                    (rv[tid + s] == rv[tid] && ri[tid + s] < ri[tid])) {
                    rv[tid] = rv[tid + s]; ri[tid] = ri[tid + s];
                }
            }
            __syncthreads();
        }
        if (tid == 0) {
            int sel_i = ri[0];
            no[k] = sel_i;
            vo[k] = (key[sel_i] <= RADIUS2) ? 1 : 0;
            key[sel_i] = 3e30f;   // exclude; stays above 1e10 so invalids keep order
        }
        __syncthreads();
    }
}

// ---------------------------------------------------------------------------
// 4) Input projection: x = [color, coord] @ W_in + b_in  (f32 + f16 copies)
// ---------------------------------------------------------------------------
__global__ void feat_kernel(const float* __restrict__ coord,
                            const float* __restrict__ color,
                            const float* __restrict__ W_in,
                            const float* __restrict__ b_in,
                            float* __restrict__ x,
                            _Float16* __restrict__ xh) {
    int t = blockIdx.x * blockDim.x + threadIdx.x;
    if (t >= ROWS * CC) return;
    int c = t % CC;
    int row = t / CC;
    float f[6];
    f[0] = color[(size_t)row * 3 + 0];
    f[1] = color[(size_t)row * 3 + 1];
    f[2] = color[(size_t)row * 3 + 2];
    f[3] = coord[(size_t)row * 3 + 0];
    f[4] = coord[(size_t)row * 3 + 1];
    f[5] = coord[(size_t)row * 3 + 2];
    float acc = b_in[c];
    #pragma unroll
    for (int j = 0; j < 6; ++j) acc += f[j] * W_in[j * CC + c];
    x[t] = acc;
    xh[t] = (_Float16)acc;
}

// ---------------------------------------------------------------------------
// Weight pre-pack into WMMA B-fragment layout (f16).
// Packed layout: [(kc * tilesN + tn) * 32 + lane] * 16 + j  -> 16 contiguous
// halfs per lane (one 32-byte load in the GEMM):
//   lanes 0-15 : col = tn*16+lane,    K = kc*32 + j        (j = 0..15)
//   lanes 16-31: col = tn*16+lane-16, K = kc*32 + 16 + j
// ---------------------------------------------------------------------------
template <int NDIM, int KDIM>
__global__ void pack_weight_kernel(const float* __restrict__ W,
                                   _Float16* __restrict__ Wp) {
    constexpr int tilesN = NDIM / 16;
    int t = blockIdx.x * blockDim.x + threadIdx.x;
    if (t >= NDIM * KDIM) return;
    int j    = t & 15;
    int lane = (t >> 4) & 31;
    int rest = t >> 9;
    int tn = rest % tilesN;
    int kc = rest / tilesN;
    int col = tn * 16 + (lane & 15);
    int kdx = kc * 32 + (lane >> 4) * 16 + j;
    Wp[t] = (_Float16)W[(size_t)kdx * NDIM + col];
}

// ---------------------------------------------------------------------------
// 5) WMMA GEMM: C[ROWS,N] = A[ROWS,K] @ W[K,N] (+bias, +relu).
//    One wave per 16x16 tile, K consumed in 32-wide v_wmma_f32_16x16x32_f16
//    steps. A: two aligned 16B vector loads per lane (row-major activations).
//    B: one contiguous 32B fragment load per lane (pre-packed weights).
//    Epilogue configuration is fully static -> branch-free stores.
// ---------------------------------------------------------------------------
template <int NDIM, int KDIM, bool HAS_BIAS, bool RELU, bool WRITE_F, bool WRITE_H>
__global__ void gemm_wmma_kernel(const _Float16* __restrict__ A,
                                 const _Float16* __restrict__ Wp,
                                 const float* __restrict__ bias,
                                 float* __restrict__ Cf,
                                 _Float16* __restrict__ Ch) {
    constexpr int tilesN = NDIM / 16;
    constexpr int tilesM = ROWS / 16;
    int wave = blockIdx.x * (blockDim.x >> 5) + (threadIdx.x >> 5);
    if (wave >= tilesM * tilesN) return;         // uniform per wave: EXEC stays full
    int lane = threadIdx.x & 31;
    int tm = wave / tilesN, tn = wave % tilesN;
    int r  = lane & 15;
    int hi = lane >> 4;                          // lane group 0..15 / 16..31

    union { v8f v; float e[8]; } c;
    #pragma unroll
    for (int j = 0; j < 8; ++j) c.e[j] = 0.0f;

    const _Float16* arow = A + (size_t)(tm * 16 + r) * KDIM;

    #pragma unroll
    for (int kc = 0; kc < KDIM / 32; ++kc) {
        int kk = kc * 32;
        // A 16x32 f16 fragment: lanes 0-15 K = kk+0..15, lanes 16-31 K = kk+16..31 (two 16B-aligned 8-half chunks).
        union { v16h v; v8h h[2]; } a;
        a.h[0] = *reinterpret_cast<const v8h*>(arow + kk + hi * 16);
        a.h[1] = *reinterpret_cast<const v8h*>(arow + kk + hi * 16 + 8);
        // B 32x16 f16 fragment: one contiguous 32B chunk from packed weights.
        v16h bfrag = *reinterpret_cast<const v16h*>(
            Wp + (((size_t)kc * tilesN + tn) * 32 + lane) * 16);

        c.v = __builtin_amdgcn_wmma_f32_16x16x32_f16(
            /*neg_a=*/false, a.v, /*neg_b=*/false, bfrag,
            /*c_mod=*/(short)0, c.v, /*reuse_a=*/false, /*reuse_b=*/false);
    }

    int col = tn * 16 + r;
    float bv = HAS_BIAS ? bias[col] : 0.0f;
    #pragma unroll
    for (int r2 = 0; r2 < 8; ++r2) {
        int row = tm * 16 + hi * 8 + r2;         // C/D layout: VGPRr -> M=r / M=8+r
        float v = c.e[r2] + bv;
        if (RELU) v = fmaxf(v, 0.0f);
        if (WRITE_F) Cf[(size_t)row * NDIM + col] = v;
        if (WRITE_H) Ch[(size_t)row * NDIM + col] = (_Float16)v;
    }
}

// ---------------------------------------------------------------------------
// 6) Neighborhood attention: one thread per (b,p,head).
//    pe = (center - neighbor) @ Wpos recomputed on the fly (cheap: 3 MACs/elt).
// ---------------------------------------------------------------------------
__global__ void attn_kernel(const float* __restrict__ q,
                            const float* __restrict__ kbuf,
                            const float* __restrict__ vbuf,
                            const float* __restrict__ coord,
                            const float* __restrict__ Wpos,
                            const int* __restrict__ nidx,
                            const int* __restrict__ nvalid,
                            float* __restrict__ ao,
                            _Float16* __restrict__ aoh) {
    int t = blockIdx.x * blockDim.x + threadIdx.x;
    if (t >= ROWS * HH) return;
    int h = t % HH;
    int row = t / HH;                 // b*NPT + p
    int b = row / NPT;

    float qr[DHD];
    const float* qp = q + (size_t)row * CC + h * DHD;
    #pragma unroll
    for (int d = 0; d < DHD; ++d) qr[d] = qp[d];

    float cx = coord[(size_t)row * 3 + 0];
    float cy = coord[(size_t)row * 3 + 1];
    float cz = coord[(size_t)row * 3 + 2];

    const int* ni = nidx + (size_t)row * KNB;
    const int* nv = nvalid + (size_t)row * KNB;

    float logits[KNB];
    float mx = -3e38f;
    for (int k = 0; k < KNB; ++k) {
        int nb = ni[k];
        size_t nrow = (size_t)b * NPT + nb;
        float dx = cx - coord[nrow * 3 + 0];
        float dy = cy - coord[nrow * 3 + 1];
        float dz = cz - coord[nrow * 3 + 2];
        const float* krow = kbuf + nrow * CC + h * DHD;
        float acc = 0.0f;
        #pragma unroll
        for (int d = 0; d < DHD; ++d) {
            int cch = h * DHD + d;
            float pe = dx * Wpos[0 * CC + cch] + dy * Wpos[1 * CC + cch] + dz * Wpos[2 * CC + cch];
            acc += qr[d] * (krow[d] + pe);
        }
        float lg = nv[k] ? (acc * 0.25f) : -1e9f;     // 1/sqrt(16)
        logits[k] = lg;
        mx = fmaxf(mx, lg);
    }
    float denom = 0.0f;
    for (int k = 0; k < KNB; ++k) {
        float e = __expf(logits[k] - mx);
        logits[k] = e;
        denom += e;
    }
    float inv = 1.0f / denom;

    float out[DHD];
    #pragma unroll
    for (int d = 0; d < DHD; ++d) out[d] = 0.0f;
    for (int k = 0; k < KNB; ++k) {
        int nb = ni[k];
        size_t nrow = (size_t)b * NPT + nb;
        float w = logits[k] * inv;
        float dx = cx - coord[nrow * 3 + 0];
        float dy = cy - coord[nrow * 3 + 1];
        float dz = cz - coord[nrow * 3 + 2];
        const float* vrow = vbuf + nrow * CC + h * DHD;
        #pragma unroll
        for (int d = 0; d < DHD; ++d) {
            int cch = h * DHD + d;
            float pe = dx * Wpos[0 * CC + cch] + dy * Wpos[1 * CC + cch] + dz * Wpos[2 * CC + cch];
            out[d] += w * (vrow[d] + pe);
        }
    }
    float* op = ao + (size_t)row * CC + h * DHD;
    _Float16* oh = aoh + (size_t)row * CC + h * DHD;
    #pragma unroll
    for (int d = 0; d < DHD; ++d) { op[d] = out[d]; oh[d] = (_Float16)out[d]; }
}

// ---------------------------------------------------------------------------
// 7) Residual + LayerNorm: one wave32 per row (C = 96 = 3 * 32 lanes).
//    Static output selection: row-major f32 + f16, or transposed (B,C,NPT).
// ---------------------------------------------------------------------------
template <bool WRITE_FH, bool WRITE_T>
__global__ void ln_kernel(const float* __restrict__ X,
                          const float* __restrict__ Add,
                          const float* __restrict__ g,
                          const float* __restrict__ be,
                          float* __restrict__ outF,
                          _Float16* __restrict__ outH,
                          float* __restrict__ outT) {
    int wid = blockIdx.x * (blockDim.x >> 5) + (threadIdx.x >> 5);
    int lane = threadIdx.x & 31;
    if (wid >= ROWS) return;
    const float* xr = X + (size_t)wid * CC;
    const float* ar = Add + (size_t)wid * CC;

    float vloc[3];
    float s = 0.0f, s2 = 0.0f;
    #pragma unroll
    for (int j = 0; j < 3; ++j) {
        float tv = xr[lane + 32 * j] + ar[lane + 32 * j];
        vloc[j] = tv; s += tv; s2 += tv * tv;
    }
    #pragma unroll
    for (int m = 16; m > 0; m >>= 1) {
        s  += __shfl_xor(s,  m, 32);
        s2 += __shfl_xor(s2, m, 32);
    }
    float mean = s * (1.0f / CC);
    float var  = s2 * (1.0f / CC) - mean * mean;
    float rs = rsqrtf(var + 1e-5f);

    int b = wid / NPT, p = wid % NPT;
    #pragma unroll
    for (int j = 0; j < 3; ++j) {
        int cidx = lane + 32 * j;
        float y = (vloc[j] - mean) * rs * g[cidx] + be[cidx];
        if (WRITE_FH) {
            outF[(size_t)wid * CC + cidx] = y;
            outH[(size_t)wid * CC + cidx] = (_Float16)y;
        }
        if (WRITE_T) {
            outT[((size_t)b * CC + cidx) * NPT + p] = y;
        }
    }
}

// ---------------------------------------------------------------------------
// Host-side orchestration
// ---------------------------------------------------------------------------
extern "C" void kernel_launch(void* const* d_in, const int* in_sizes, int n_in,
                              void* d_out, int out_size, void* d_ws, size_t ws_size,
                              hipStream_t stream) {
    const float* point_clouds = (const float*)d_in[0];
    const float* cloud_colors = (const float*)d_in[1];
    const float* W_in = (const float*)d_in[2];
    const float* b_in = (const float*)d_in[3];
    const float* Wq   = (const float*)d_in[4];
    const float* Wk   = (const float*)d_in[5];
    const float* Wv   = (const float*)d_in[6];
    const float* Wo   = (const float*)d_in[7];
    const float* Wpos = (const float*)d_in[8];
    const float* W1   = (const float*)d_in[9];
    const float* b1   = (const float*)d_in[10];
    const float* W2   = (const float*)d_in[11];
    const float* b2   = (const float*)d_in[12];
    const float* g1   = (const float*)d_in[13];
    const float* be1  = (const float*)d_in[14];
    const float* g2   = (const float*)d_in[15];
    const float* be2  = (const float*)d_in[16];
    const int*   sel  = (const int*)d_in[17];

    float* out = (float*)d_out;
    float* outFeatures = out;                          // B*C*NPT
    float* outCoord    = out + (size_t)BB * CC * NPT;  // B*NPT*3

    char* ws = (char*)d_ws;
    size_t off = 0;
    auto alloc = [&](size_t bytes) -> void* {
        void* p = ws + off;
        off = (off + bytes + 255) & ~(size_t)255;
        return p;
    };

    float* minDist = (float*)alloc((size_t)BB * N_IN * sizeof(float));
    int*   fpsIdx  = (int*)  alloc((size_t)BB * N_ALL * sizeof(int));
    float* coord   = (float*)alloc((size_t)ROWS * 3 * sizeof(float));
    float* color   = (float*)alloc((size_t)ROWS * 3 * sizeof(float));
    int*   nidx    = (int*)  alloc((size_t)ROWS * KNB * sizeof(int));
    int*   nvalid  = (int*)  alloc((size_t)ROWS * KNB * sizeof(int));
    float* x       = (float*)alloc((size_t)ROWS * CC * sizeof(float));
    _Float16* xh   = (_Float16*)alloc((size_t)ROWS * CC * sizeof(_Float16));
    float* qbuf    = (float*)alloc((size_t)ROWS * CC * sizeof(float));
    float* kbuf    = (float*)alloc((size_t)ROWS * CC * sizeof(float));
    float* vbuf    = (float*)alloc((size_t)ROWS * CC * sizeof(float));
    float* ao      = (float*)alloc((size_t)ROWS * CC * sizeof(float));
    _Float16* aoh  = (_Float16*)alloc((size_t)ROWS * CC * sizeof(_Float16));
    float* obuf    = (float*)alloc((size_t)ROWS * CC * sizeof(float));
    float* x1      = (float*)alloc((size_t)ROWS * CC * sizeof(float));
    _Float16* x1h  = (_Float16*)alloc((size_t)ROWS * CC * sizeof(_Float16));
    _Float16* h1h  = (_Float16*)alloc((size_t)ROWS * 4 * CC * sizeof(_Float16));
    float* h2      = (float*)alloc((size_t)ROWS * CC * sizeof(float));
    _Float16* Wq_p = (_Float16*)alloc((size_t)CC * CC * sizeof(_Float16));
    _Float16* Wk_p = (_Float16*)alloc((size_t)CC * CC * sizeof(_Float16));
    _Float16* Wv_p = (_Float16*)alloc((size_t)CC * CC * sizeof(_Float16));
    _Float16* Wo_p = (_Float16*)alloc((size_t)CC * CC * sizeof(_Float16));
    _Float16* W1_p = (_Float16*)alloc((size_t)CC * 4 * CC * sizeof(_Float16));
    _Float16* W2_p = (_Float16*)alloc((size_t)4 * CC * CC * sizeof(_Float16));
    (void)ws_size; (void)in_sizes; (void)n_in; (void)out_size;

    // 1) FPS
    fps_kernel<<<BB, 1024, 0, stream>>>(point_clouds, minDist, fpsIdx);

    // 2) Gather selected points
    gather_kernel<<<(ROWS + 255) / 256, 256, 0, stream>>>(
        point_clouds, cloud_colors, fpsIdx, sel, coord, color, outCoord);

    // 3) kNN selection
    knn_kernel<<<ROWS, 256, 0, stream>>>(coord, nidx, nvalid);

    // 4) Weight packing into WMMA fragment layout (f16)
    pack_weight_kernel<CC, CC><<<(CC * CC + 255) / 256, 256, 0, stream>>>(Wq, Wq_p);
    pack_weight_kernel<CC, CC><<<(CC * CC + 255) / 256, 256, 0, stream>>>(Wk, Wk_p);
    pack_weight_kernel<CC, CC><<<(CC * CC + 255) / 256, 256, 0, stream>>>(Wv, Wv_p);
    pack_weight_kernel<CC, CC><<<(CC * CC + 255) / 256, 256, 0, stream>>>(Wo, Wo_p);
    pack_weight_kernel<4 * CC, CC><<<(CC * 4 * CC + 255) / 256, 256, 0, stream>>>(W1, W1_p);
    pack_weight_kernel<CC, 4 * CC><<<(4 * CC * CC + 255) / 256, 256, 0, stream>>>(W2, W2_p);

    // 5) Input projection
    feat_kernel<<<(ROWS * CC + 255) / 256, 256, 0, stream>>>(
        coord, color, W_in, b_in, x, xh);

    // 6) q/k/v projections via WMMA (tiles = 512*6; 8 waves/block)
    constexpr int blk96  = (ROWS / 16) * (CC / 16) / 8;        // 384
    constexpr int blk384 = (ROWS / 16) * (4 * CC / 16) / 8;    // 1536
    gemm_wmma_kernel<CC, CC, false, false, true, false>
        <<<blk96, 256, 0, stream>>>(xh, Wq_p, nullptr, qbuf, nullptr);
    gemm_wmma_kernel<CC, CC, false, false, true, false>
        <<<blk96, 256, 0, stream>>>(xh, Wk_p, nullptr, kbuf, nullptr);
    gemm_wmma_kernel<CC, CC, false, false, true, false>
        <<<blk96, 256, 0, stream>>>(xh, Wv_p, nullptr, vbuf, nullptr);

    // 7) Neighborhood attention
    attn_kernel<<<(ROWS * HH + 255) / 256, 256, 0, stream>>>(
        qbuf, kbuf, vbuf, coord, Wpos, nidx, nvalid, ao, aoh);

    // 8) Output projection o = attn_out @ Wo, then LN1
    gemm_wmma_kernel<CC, CC, false, false, true, false>
        <<<blk96, 256, 0, stream>>>(aoh, Wo_p, nullptr, obuf, nullptr);
    ln_kernel<true, false><<<(ROWS + 7) / 8, 256, 0, stream>>>(
        x, obuf, g1, be1, x1, x1h, nullptr);

    // 9) FFN: h1 = relu(x1 @ W1 + b1) [f16 only], h2 = h1 @ W2 + b2
    gemm_wmma_kernel<4 * CC, CC, true, true, false, true>
        <<<blk384, 256, 0, stream>>>(x1h, W1_p, b1, nullptr, h1h);
    gemm_wmma_kernel<CC, 4 * CC, true, false, true, false>
        <<<blk96, 256, 0, stream>>>(h1h, W2_p, b2, h2, nullptr);

    // 10) LN2 + transpose write to output features (B, C, NPT)
    ln_kernel<false, true><<<(ROWS + 7) / 8, 256, 0, stream>>>(
        x1, h2, g2, be2, nullptr, nullptr, outFeatures);
}
